// SparseExpertModule_69209103007761
// MI455X (gfx1250) — matrix operand order repaired and checked by reading, not compiled, not run on hardware
//
#include <hip/hip_runtime.h>
#include <hip/hip_bf16.h>

typedef __attribute__((ext_vector_type(16))) _Float16 v16h;
typedef __attribute__((ext_vector_type(8)))  float    v8f;

#define TOK   4096   // B*S
#define HDIM  1024
#define NEXP  8
#define FDIM  4096
#define FCHK  128    // F chunk per outer iteration
#define TROWS 32     // tokens per block (two 16-row WMMA tiles)

// fragments: 512 halfs (1 KB) each; lane's 16 halfs contiguous (32 B)
#define NF1_PER_E  ((FDIM / 16) * (HDIM / 32))   // 8192 W1 frags / expert
#define NF2_PER_E  ((HDIM / 16) * (FDIM / 32))   // 8192 W2 frags / expert
#define NF1        (NEXP * NF1_PER_E)            // 65536
#define NF2        (NEXP * NF2_PER_E)            // 65536

// ---------------------------------------------------------------- zero + counts
__global__ void zero_out_kernel(float* __restrict__ out, int n4, int* __restrict__ counts) {
    int i = blockIdx.x * blockDim.x + threadIdx.x;
    if (i < n4) {
        float4 z; z.x = 0.f; z.y = 0.f; z.z = 0.f; z.w = 0.f;
        ((float4*)out)[i] = z;
    }
    if (blockIdx.x == 0 && threadIdx.x < NEXP) counts[threadIdx.x] = 0;
}

// ---------------------------------------------------------------- router
__global__ void router_kernel(const float* __restrict__ h, const float* __restrict__ Wr,
                              int* __restrict__ counts, int* __restrict__ lists,
                              float* __restrict__ gates) {
    const int wave  = threadIdx.x >> 5;
    const int lane  = threadIdx.x & 31;
    const int token = blockIdx.x * 8 + wave;

    float acc[NEXP];
#pragma unroll
    for (int e = 0; e < NEXP; ++e) acc[e] = 0.f;

    const float* hrow = h + (size_t)token * HDIM;
    for (int i = lane; i < HDIM; i += 32) {
        const float  hv = hrow[i];
        const float* wr = Wr + i * NEXP;
#pragma unroll
        for (int e = 0; e < NEXP; ++e) acc[e] = fmaf(hv, wr[e], acc[e]);
    }
#pragma unroll
    for (int e = 0; e < NEXP; ++e) {
#pragma unroll
        for (int off = 16; off > 0; off >>= 1)
            acc[e] += __shfl_xor(acc[e], off, 32);
    }

    if (lane == 0) {
        int   i0 = 0; float l0 = acc[0];
#pragma unroll
        for (int e = 1; e < NEXP; ++e) if (acc[e] > l0) { l0 = acc[e]; i0 = e; }
        int   i1 = -1; float l1 = -3.4e38f;
#pragma unroll
        for (int e = 0; e < NEXP; ++e) if (e != i0 && acc[e] > l1) { l1 = acc[e]; i1 = e; }
        const float w0 = 1.f / (1.f + __expf(l1 - l0));  // renormalized top-2 softmax
        const float w1 = 1.f - w0;
        gates[token * 2 + 0] = w0;
        gates[token * 2 + 1] = w1;
        int p0 = atomicAdd(&counts[i0], 1);
        lists[i0 * TOK + p0] = token * 2 + 0;
        int p1 = atomicAdd(&counts[i1], 1);
        lists[i1 * TOK + p1] = token * 2 + 1;
    }
}

// ---------------------------------------------------------------- B fragment from raw f32 W (fallback path)
__device__ __forceinline__ v16h load_b_g(const float* __restrict__ W, int ld,
                                         int kbase, int nbase, int N, int hk) {
    v16h b;
#pragma unroll
    for (int v = 0; v < 8; ++v) {
        const int kk = kbase + hk * 16 + 2 * v;
        b[2 * v]     = (_Float16)W[(size_t)kk * ld + nbase + N];
        b[2 * v + 1] = (_Float16)W[(size_t)(kk + 1) * ld + nbase + N];
    }
    return b;
}

// A-fragment packing: element (row r<16, K-local kl<32) -> (lane, slot)
__device__ __forceinline__ void a_pack_pos(int r, int kl, int& lane_t, int& slot) {
    const int hkt = kl >> 4;               // lane half: 0 -> K0-15, 1 -> K16-31 (contiguous)
    slot   = kl & 15;
    lane_t = r + 16 * hkt;
}

// ---------------------------------------------------------------- weight pre-pack (f32 -> f16 fragment layout)
__global__ __launch_bounds__(256)
void prepack_kernel(const float* __restrict__ W1, const float* __restrict__ W2,
                    _Float16* __restrict__ pW1, _Float16* __restrict__ pW2) {
    const int wave = threadIdx.x >> 5, lane = threadIdx.x & 31;
    const int N = lane & 15, hk = lane >> 4;
    const long fid = (long)blockIdx.x * 8 + wave;

    const float* src; _Float16* dst; int ld, kb, nb;
    if (fid < NF1) {
        const int e = (int)(fid >> 13), r = (int)(fid & 8191);
        const int ntile = r >> 5, ktile = r & 31;          // 256 ntiles x 32 ktiles
        src = W1 + (size_t)e * HDIM * FDIM; ld = FDIM;
        kb = ktile * 32; nb = ntile * 16;
        dst = pW1 + (size_t)fid * 512;
    } else {
        const long f2 = fid - NF1;
        const int e = (int)(f2 >> 13), r = (int)(f2 & 8191);
        const int ntile = r >> 7, ktile = r & 127;         // 64 ntiles x 128 ktiles
        src = W2 + (size_t)e * FDIM * HDIM; ld = HDIM;
        kb = ktile * 32; nb = ntile * 16;
        dst = pW2 + (size_t)f2 * 512;
    }
    v16h b;
#pragma unroll
    for (int v = 0; v < 8; ++v) {
        const int kk = kb + hk * 16 + 2 * v;
        b[2 * v]     = (_Float16)src[(size_t)kk * ld + nb + N];
        b[2 * v + 1] = (_Float16)src[(size_t)(kk + 1) * ld + nb + N];
    }
    *(v16h*)(dst + (size_t)lane * 16) = b;
}

// ---------------------------------------------------------------- fused expert FFN
// 256 threads = 8 wave32; one 32-token tile of one expert (two 16-row WMMA row-sets).
template <bool PACKED>
__global__ __launch_bounds__(256)
void moe_ffn_kernel(const float* __restrict__ h,
                    const float* __restrict__ W1, const float* __restrict__ W2,
                    const _Float16* __restrict__ pW1, const _Float16* __restrict__ pW2,
                    const int* __restrict__ counts, const int* __restrict__ lists,
                    const float* __restrict__ gates, float* __restrict__ out) {
    __shared__ __align__(16) float    sStage[4 * HDIM];               // 16 KB async dest (4 rows)
    __shared__ __align__(32) _Float16 sAp[(HDIM / 32) * 2 * 512];     // 64 KB packed A frags, 2 row-sets
    __shared__ __align__(32) _Float16 sHp[(FCHK / 32) * 2 * 512];     //  8 KB packed hid frags
    __shared__ int   sTok[TROWS];
    __shared__ float sGate[TROWS];

    const int e    = blockIdx.y;
    const int tile = blockIdx.x;
    const int cnt  = counts[e];
    if (tile * TROWS >= cnt) return;

    const int tid  = threadIdx.x;
    const int wave = tid >> 5;
    const int lane = tid & 31;
    const int M    = lane & 15;
    const int hk   = lane >> 4;

    if (tid < TROWS) {
        const int gi = tile * TROWS + tid;
        if (gi < cnt) {
            const int en = lists[e * TOK + gi];
            sTok[tid]  = en >> 1;
            sGate[tid] = gates[en];
        } else {
            sTok[tid]  = 0;
            sGate[tid] = 0.f;
        }
    }
    __syncthreads();

    // ---- gather 32 h rows via async global->LDS DMA, convert+pack to A-fragment layout
    const unsigned sStageBase = (unsigned)(uintptr_t)&sStage[0];
    for (int g = 0; g < TROWS / 4; ++g) {         // 4 rows per round
        for (int q = tid; q < 1024; q += 256) {   // 1024 float4 = 16 KB
            const int r  = g * 4 + (q >> 8);
            const int c4 = q & 255;
            const float* gp = h + (size_t)sTok[r] * HDIM + (size_t)c4 * 4;
            const unsigned lo = sStageBase + (unsigned)q * 16u;
            asm volatile("global_load_async_to_lds_b128 %0, %1, off"
                         :: "v"(lo), "v"(gp) : "memory");
        }
        asm volatile("s_wait_asynccnt 0x0" ::: "memory");
        __syncthreads();
        for (int idx = tid; idx < 4096; idx += 256) {
            const int r = g * 4 + (idx >> 10);
            const int c = idx & (HDIM - 1);
            const float val = sStage[idx];
            const int set = r >> 4, rr = r & 15;
            int lane_t, slot;
            a_pack_pos(rr, c & 31, lane_t, slot);
            sAp[(c >> 5) * 1024 + set * 512 + lane_t * 16 + slot] = (_Float16)val;
        }
        __syncthreads();
    }

    const float* W1e = W1 + (size_t)e * HDIM * FDIM;
    const float* W2e = W2 + (size_t)e * FDIM * HDIM;

    v8f acc[16];                                   // [set][t]: 2 row-sets x 8 col tiles
#pragma unroll
    for (int t = 0; t < 16; ++t) acc[t] = (v8f){0.f,0.f,0.f,0.f,0.f,0.f,0.f,0.f};

    for (int fc = 0; fc < FDIM; fc += FCHK) {
        // prefetch next packed-W2 chunk for this wave
        if (PACKED && fc + FCHK < FDIM) {
            const _Float16* nf = pW2 + ((size_t)e * NF2_PER_E
                                        + (size_t)(wave * 8) * 128 + ((fc + FCHK) >> 5)) * 512;
            for (int q = lane; q < 256; q += 32) {        // 8 ntiles x 4 ktiles x 128B segs
                const int t = q >> 5, rest = q & 31, ktl = rest >> 3, seg = rest & 7;
                __builtin_prefetch(nf + ((size_t)t * 128 + ktl) * 512 + seg * 64, 0, 1);
            }
        }

        // ---- stage 1: hid chunk = relu(h_tile @ W1[:, fc:fc+128]); wave owns 16 cols, 32 rows
        v8f ch0 = (v8f){0.f,0.f,0.f,0.f,0.f,0.f,0.f,0.f};
        v8f ch1 = ch0;
        const int nb = fc + wave * 16;
        if (PACKED) {
            const _Float16* fb = pW1 + ((size_t)e * NF1_PER_E + (size_t)(nb >> 4) * 32) * 512
                                     + (size_t)lane * 16;
            v16h b_cur = *(const v16h*)fb;                       // software-pipelined B
            for (int kt = 0; kt < HDIM / 32; ++kt) {
                v16h a0 = *(const v16h*)&sAp[kt * 1024 + lane * 16];
                v16h a1 = *(const v16h*)&sAp[kt * 1024 + 512 + lane * 16];
                const v16h b = b_cur;
                if (kt < HDIM / 32 - 1) b_cur = *(const v16h*)(fb + (size_t)(kt + 1) * 512);
                ch0 = __builtin_amdgcn_wmma_f32_16x16x32_f16(false, a0, false, b,
                                                             (short)0, ch0, false, false);
                ch1 = __builtin_amdgcn_wmma_f32_16x16x32_f16(false, a1, false, b,
                                                             (short)0, ch1, false, false);
            }
        } else {
            for (int kb = 0; kb < HDIM; kb += 32) {
                v16h a0 = *(const v16h*)&sAp[(kb >> 5) * 1024 + lane * 16];
                v16h a1 = *(const v16h*)&sAp[(kb >> 5) * 1024 + 512 + lane * 16];
                v16h b  = load_b_g(W1e, FDIM, kb, nb, M, hk);
                ch0 = __builtin_amdgcn_wmma_f32_16x16x32_f16(false, a0, false, b,
                                                             (short)0, ch0, false, false);
                ch1 = __builtin_amdgcn_wmma_f32_16x16x32_f16(false, a1, false, b,
                                                             (short)0, ch1, false, false);
            }
        }
        __syncthreads();                       // previous stage-2 reads of sHp done
        {
            // D layout: VGPR i -> row i (lanes 0-15) / i+8 (lanes 16-31), col = wave*16+M
            const int kl = ((wave & 1) << 4) + M;
            int lane_t, slot;
            a_pack_pos(0, kl, lane_t, slot);   // row added below
            const int kt = wave >> 1;
#pragma unroll
            for (int i = 0; i < 8; ++i) {
                const int row = i + 8 * hk;
                sHp[kt * 1024 + (lane_t + row) * 16 + slot]       = (_Float16)fmaxf(ch0[i], 0.f);
                sHp[kt * 1024 + 512 + (lane_t + row) * 16 + slot] = (_Float16)fmaxf(ch1[i], 0.f);
            }
        }
        __syncthreads();

        // ---- stage 2: acc += hid_chunk @ W2[fc:fc+128, wave's 128 cols]
        for (int kk = 0; kk < FCHK; kk += 32) {
            const int kt = kk >> 5;
            v16h a20 = *(const v16h*)&sHp[kt * 1024 + lane * 16];
            v16h a21 = *(const v16h*)&sHp[kt * 1024 + 512 + lane * 16];
            if (PACKED) {
                const _Float16* fb2 = pW2 + ((size_t)e * NF2_PER_E
                                             + (size_t)(wave * 8) * 128 + ((fc + kk) >> 5)) * 512
                                          + (size_t)lane * 16;
#pragma unroll
                for (int t = 0; t < 8; ++t) {
                    v16h b2 = *(const v16h*)(fb2 + (size_t)t * 128 * 512);
                    acc[t]     = __builtin_amdgcn_wmma_f32_16x16x32_f16(false, a20, false, b2,
                                                                        (short)0, acc[t], false, false);
                    acc[8 + t] = __builtin_amdgcn_wmma_f32_16x16x32_f16(false, a21, false, b2,
                                                                        (short)0, acc[8 + t], false, false);
                }
            } else {
#pragma unroll
                for (int t = 0; t < 8; ++t) {
                    v16h b2 = load_b_g(W2e, HDIM, fc + kk, wave * 128 + t * 16, M, hk);
                    acc[t]     = __builtin_amdgcn_wmma_f32_16x16x32_f16(false, a20, false, b2,
                                                                        (short)0, acc[t], false, false);
                    acc[8 + t] = __builtin_amdgcn_wmma_f32_16x16x32_f16(false, a21, false, b2,
                                                                        (short)0, acc[8 + t], false, false);
                }
            }
        }
    }

    // ---- epilogue: scale by gate; exactly 2 commutative adds per element -> deterministic
#pragma unroll
    for (int t = 0; t < 8; ++t) {
#pragma unroll
        for (int i = 0; i < 8; ++i) {
            const int row = i + 8 * hk;
            const int col = wave * 128 + t * 16 + M;
            const float v0 = acc[t][i]     * sGate[row];
            const float v1 = acc[8 + t][i] * sGate[16 + row];
            atomicAdd(out + (size_t)sTok[row] * HDIM + col, v0);
            atomicAdd(out + (size_t)sTok[16 + row] * HDIM + col, v1);
        }
    }
}

// ---------------------------------------------------------------- launch
extern "C" void kernel_launch(void* const* d_in, const int* in_sizes, int n_in,
                              void* d_out, int out_size, void* d_ws, size_t ws_size,
                              hipStream_t stream) {
    const float* h  = (const float*)d_in[0];   // [2,2048,1024]
    const float* Wr = (const float*)d_in[1];   // [1024,8]
    const float* W1 = (const float*)d_in[2];   // [8,1024,4096]
    const float* W2 = (const float*)d_in[3];   // [8,4096,1024]
    float* out = (float*)d_out;                // [2,2048,1024]

    // meta workspace: counts[8] | lists[8][4096] | gates[4096*2]
    int*   counts = (int*)d_ws;
    int*   lists  = (int*)((char*)d_ws + 128);
    float* gates  = (float*)((char*)d_ws + 128 + (size_t)NEXP * TOK * sizeof(int));

    const size_t META       = (size_t)1 << 20;
    const size_t PACK_HALFS = (size_t)2 * NEXP * HDIM * FDIM;     // W1h + W2h
    _Float16* pW1 = (_Float16*)((char*)d_ws + META);
    _Float16* pW2 = pW1 + (size_t)NEXP * HDIM * FDIM;
    const bool packed = ws_size >= META + PACK_HALFS * sizeof(_Float16);

    const int n = TOK * HDIM;
    zero_out_kernel<<<(n / 4 + 255) / 256, 256, 0, stream>>>(out, n / 4, counts);
    router_kernel<<<TOK / 8, 256, 0, stream>>>(h, Wr, counts, lists, gates);

    if (packed) {
        prepack_kernel<<<(NF1 + NF2) / 8, 256, 0, stream>>>(W1, W2, pW1, pW2);
        moe_ffn_kernel<true><<<dim3(TOK / TROWS, NEXP), 256, 0, stream>>>(
            h, W1, W2, pW1, pW2, counts, lists, gates, out);
    } else {
        moe_ffn_kernel<false><<<dim3(TOK / TROWS, NEXP), 256, 0, stream>>>(
            h, W1, W2, pW1, pW2, counts, lists, gates, out);
    }
}
